// AIGStateEncoder_27831388078623
// MI455X (gfx1250) — matrix operand order repaired and checked by reading, not compiled, not run on hardware
//
#include <hip/hip_runtime.h>
#include <hip/hip_bf16.h>

typedef __attribute__((ext_vector_type(16))) _Float16 v16h;
typedef __attribute__((ext_vector_type(8)))  _Float16 v8h;
typedef __attribute__((ext_vector_type(8)))  float    v8f;
typedef __attribute__((ext_vector_type(4)))  float    v4f;

#define TSEQ 256
#define HD   64

// ---- branch-free transcendentals (keep EXEC untouched in the hot loop) ----
__device__ __forceinline__ float fast_sigmoid(float x) {
    // v_exp_f32 + v_rcp_f32, no range-split branches
    return __builtin_amdgcn_rcpf(1.0f + __expf(-x));
}

__device__ __forceinline__ float fast_tanh(float x) {
#if __has_builtin(__builtin_amdgcn_tanhf)
    return __builtin_amdgcn_tanhf(x);          // v_tanh_f32 (CDNA5 TRANS op)
#else
    float e = __expf(2.0f * x);
    return 1.0f - 2.0f * __builtin_amdgcn_rcpf(e + 1.0f);
#endif
}

// B-fragment (32x16, f16) for V_WMMA_F32_16X16X32_F16, built from an fp32
// matrix W stored [N][K] with K-stride 64 (i.e. B[k][n] = W[n][k]).
// Lanes 0-15 hold K=Kb..Kb+7 and Kb+16..Kb+23 at column N; lanes 16-31 add +8 (Kb folds it).
__device__ __forceinline__ v16h load_bfrag_f32(const float* __restrict__ W, int N, int Kb) {
    v16h r;
    const float* p = W + N * HD + Kb;
#pragma unroll
    for (int j = 0; j < 16; ++j) r[j] = (_Float16)p[j < 8 ? j : j + 8];
    return r;
}

// A-fragment (16x32, f16) from an LDS tile h[16][64].
// Lane's row m = lane&15; hi = lane>>4 selects the +8 / +24 K-chunks.
__device__ __forceinline__ v16h load_afrag_lds(const _Float16* hbase, int m, int kk, int hi) {
    const _Float16* p = hbase + m * HD + 32 * kk + 8 * hi;
    v8h lo = *(const v8h*)(p);
    v8h hh = *(const v8h*)(p + 16);
    v16h a;
#pragma unroll
    for (int j = 0; j < 8; ++j) { a[j] = lo[j]; a[8 + j] = hh[j]; }
    return a;
}

__global__ __launch_bounds__(128, 1)
void lstm2_fused_wmma_kernel(const float* __restrict__ recipe,
                             const float* __restrict__ W_ih0,
                             const float* __restrict__ W_hh0,
                             const float* __restrict__ b_ih0,
                             const float* __restrict__ b_hh0,
                             const float* __restrict__ W_ih1,
                             const float* __restrict__ W_hh1,
                             const float* __restrict__ b_ih1,
                             const float* __restrict__ b_hh1,
                             const float* __restrict__ W_lin,
                             const float* __restrict__ b_lin,
                             float* __restrict__ out)
{
    __shared__ _Float16 h0buf[2][16][HD];   // layer-0 hidden, double-buffered
    __shared__ _Float16 h1buf[2][16][HD];   // layer-1 hidden, double-buffered
    __shared__ float    xbuf[TSEQ][16];     // recipe tile, [t][batch-row]

    const int tid   = threadIdx.x;
    const int lane  = tid & 31;
    const int wv    = tid >> 5;        // wave 0..3: owns h-channels [16w,16w+16)
    const int col   = lane & 15;
    const int hi    = lane >> 4;
    const int m     = col;             // A-matrix row handled by this lane
    const int nch   = 16 * wv + col;   // hidden channel index 0..63
    const int batch0 = blockIdx.x * 16;

    // ---- zero hidden-state buffers ----
    {
        _Float16* z0 = &h0buf[0][0][0];
        _Float16* z1 = &h1buf[0][0][0];
        for (int i = tid; i < 2 * 16 * HD; i += 128) {
            z0[i] = (_Float16)0.0f;
            z1[i] = (_Float16)0.0f;
        }
    }
    // ---- stage the input sequence for this batch tile (coalesced) ----
    for (int i = tid; i < 16 * TSEQ; i += 128) {
        int mr = i / TSEQ, t = i % TSEQ;
        xbuf[t][mr] = recipe[(size_t)(batch0 + mr) * TSEQ + t];
    }

    // ---- per-lane scalar params (layer-0 input contribution + biases) ----
    float wx[4], bias0v[4], bias1v[4];
#pragma unroll
    for (int g = 0; g < 4; ++g) {
        int gn = 64 * g + nch;
        wx[g]     = W_ih0[gn];                 // W_ih0 is [256][1]
        bias0v[g] = b_ih0[gn] + b_hh0[gn];
        bias1v[g] = b_ih1[gn] + b_hh1[gn];
    }

    // ---- resident f16 weight fragments: W_hh0, W_ih1, W_hh1 (each [256][64]) ----
    v16h wbhh0[4][2], wbih1[4][2], wbhh1[4][2];
#pragma unroll
    for (int g = 0; g < 4; ++g) {
#pragma unroll
        for (int kk = 0; kk < 2; ++kk) {
            int N  = 64 * g + nch;
            int Kb = 32 * kk + 8 * hi;
            wbhh0[g][kk] = load_bfrag_f32(W_hh0, N, Kb);
            wbih1[g][kk] = load_bfrag_f32(W_ih1, N, Kb);
            wbhh1[g][kk] = load_bfrag_f32(W_hh1, N, Kb);
        }
    }

    // ---- cell states (lane-local: batch rows hi*8..hi*8+7, channel nch) ----
    float c0s[8], c1s[8];
#pragma unroll
    for (int r = 0; r < 8; ++r) { c0s[r] = 0.0f; c1s[r] = 0.0f; }

    __syncthreads();

    const v8f vzero = {};

#pragma unroll 1
    for (int t = 0; t < TSEQ; ++t) {
        const int cur = t & 1, prv = cur ^ 1;

        // ================= layer 0: g = h0_prev @ W_hh0^T (+ x*W_ih0 + b) ====
        v8f acc[4];
#pragma unroll
        for (int g = 0; g < 4; ++g) acc[g] = vzero;
#pragma unroll
        for (int kk = 0; kk < 2; ++kk) {
            v16h a = load_afrag_lds(&h0buf[prv][0][0], m, kk, hi);
#pragma unroll
            for (int g = 0; g < 4; ++g)
                acc[g] = __builtin_amdgcn_wmma_f32_16x16x32_f16(
                    false, a, false, wbhh0[g][kk], (short)0, acc[g], false, false);
        }
        // x for this lane's 8 batch rows: contiguous -> two b128 LDS loads
        float xv[8];
        {
            const float* xp = &xbuf[t][8 * hi];
            v4f x0 = *(const v4f*)(xp);
            v4f x1 = *(const v4f*)(xp + 4);
#pragma unroll
            for (int j = 0; j < 4; ++j) { xv[j] = x0[j]; xv[4 + j] = x1[j]; }
        }
#pragma unroll
        for (int r = 0; r < 8; ++r) {
            float xr = xv[r];
            float gi = fast_sigmoid(acc[0][r] + xr * wx[0] + bias0v[0]);
            float gf = fast_sigmoid(acc[1][r] + xr * wx[1] + bias0v[1]);
            float gg = fast_tanh   (acc[2][r] + xr * wx[2] + bias0v[2]);
            float go = fast_sigmoid(acc[3][r] + xr * wx[3] + bias0v[3]);
            float c  = gf * c0s[r] + gi * gg;
            c0s[r]   = c;
            float h  = go * fast_tanh(c);
            h0buf[cur][r + 8 * hi][nch] = (_Float16)h;
        }
        __syncthreads();

        // ====== layer 1: g = h0_cur @ W_ih1^T + h1_prev @ W_hh1^T + b ========
#pragma unroll
        for (int g = 0; g < 4; ++g) acc[g] = vzero;
#pragma unroll
        for (int kk = 0; kk < 2; ++kk) {
            v16h ain = load_afrag_lds(&h0buf[cur][0][0], m, kk, hi);
            v16h arc = load_afrag_lds(&h1buf[prv][0][0], m, kk, hi);
#pragma unroll
            for (int g = 0; g < 4; ++g) {
                acc[g] = __builtin_amdgcn_wmma_f32_16x16x32_f16(
                    false, ain, false, wbih1[g][kk], (short)0, acc[g], false, false);
                acc[g] = __builtin_amdgcn_wmma_f32_16x16x32_f16(
                    false, arc, false, wbhh1[g][kk], (short)0, acc[g], false, false);
            }
        }
#pragma unroll
        for (int r = 0; r < 8; ++r) {
            float gi = fast_sigmoid(acc[0][r] + bias1v[0]);
            float gf = fast_sigmoid(acc[1][r] + bias1v[1]);
            float gg = fast_tanh   (acc[2][r] + bias1v[2]);
            float go = fast_sigmoid(acc[3][r] + bias1v[3]);
            float c  = gf * c1s[r] + gi * gg;
            c1s[r]   = c;
            float h  = go * fast_tanh(c);
            h1buf[cur][r + 8 * hi][nch] = (_Float16)h;
        }
        __syncthreads();
    }

    // ====== final linear: out = h1_{T-1} @ W_lin^T + b_lin  ([16,64]@[64,64]) ==
    {
        const int last = (TSEQ - 1) & 1;
        v8f acc = vzero;
#pragma unroll
        for (int kk = 0; kk < 2; ++kk) {
            v16h a = load_afrag_lds(&h1buf[last][0][0], m, kk, hi);
            v16h b = load_bfrag_f32(W_lin, nch, 32 * kk + 8 * hi);  // W_lin [64][64]
            acc = __builtin_amdgcn_wmma_f32_16x16x32_f16(
                false, a, false, b, (short)0, acc, false, false);
        }
        float bl = b_lin[nch];
#pragma unroll
        for (int r = 0; r < 8; ++r)
            out[(size_t)(batch0 + r + 8 * hi) * HD + nch] = acc[r] + bl;
    }
}

extern "C" void kernel_launch(void* const* d_in, const int* in_sizes, int n_in,
                              void* d_out, int out_size, void* d_ws, size_t ws_size,
                              hipStream_t stream) {
    const float* recipe = (const float*)d_in[0];
    const float* W_ih0  = (const float*)d_in[1];
    const float* W_hh0  = (const float*)d_in[2];
    const float* b_ih0  = (const float*)d_in[3];
    const float* b_hh0  = (const float*)d_in[4];
    const float* W_ih1  = (const float*)d_in[5];
    const float* W_hh1  = (const float*)d_in[6];
    const float* b_ih1  = (const float*)d_in[7];
    const float* b_hh1  = (const float*)d_in[8];
    const float* W_lin  = (const float*)d_in[9];
    const float* b_lin  = (const float*)d_in[10];
    float* out = (float*)d_out;

    // 4096 batch rows / 16 per workgroup = 256 workgroups, 4 waves each.
    lstm2_fused_wmma_kernel<<<256, 128, 0, stream>>>(
        recipe, W_ih0, W_hh0, b_ih0, b_hh0,
        W_ih1, W_hh1, b_ih1, b_hh1, W_lin, b_lin, out);
}
